// Decoder_9002251453267
// MI455X (gfx1250) — matrix operand order repaired and checked by reading, not compiled, not run on hardware
//
#include <hip/hip_runtime.h>
#include <hip/hip_bf16.h>
#include <stdint.h>

typedef __bf16 bf16_t;
typedef __attribute__((ext_vector_type(16))) __bf16 v16bf;
typedef __attribute__((ext_vector_type(8)))  float  v8f;

#define SLENC 64
#define BATCH 64
#define TLENC 48

static __device__ __forceinline__ uint32_t f2bf_bits(float f) {
  union { float f; uint32_t u; } x; x.f = f;
  uint32_t u = x.u;
  return ((u + 0x7FFFu + ((u >> 16) & 1u)) >> 16) & 0xFFFFu;  // RNE
}
static __device__ __forceinline__ bf16_t f2bf(float f) {
  union { unsigned short s; bf16_t b; } y;
  y.s = (unsigned short)f2bf_bits(f);
  return y.b;
}

// ---------------------------------------------------------------- utilities
__global__ void zero_u32_kernel(uint32_t* p, int n) {
  int i = blockIdx.x * 256 + threadIdx.x;
  if (i < n) p[i] = 0u;
}

// Pack fp32 weight [Ks,Ns] row-major into K-pair-packed, column-major bf16:
// dst[(colOff+n)*K2tot + rowOff2 + kp] = {bf16(W[2kp][n]), bf16(W[2kp+1][n])}
__global__ void pack_weight_kernel(const float* __restrict__ src, int Ks, int Ns,
                                   uint32_t* __restrict__ dst, int K2tot,
                                   int rowOff2, int colOff, int total) {
  int idx = blockIdx.x * 256 + threadIdx.x;
  if (idx >= total) return;
  int K2 = Ks >> 1;
  int kp = idx % K2;
  int n  = idx / K2;
  uint32_t lo = f2bf_bits(src[(size_t)(2 * kp) * Ns + n]);
  uint32_t hi = f2bf_bits(src[(size_t)(2 * kp + 1) * Ns + n]);
  dst[(size_t)(colOff + n) * K2tot + rowOff2 + kp] = lo | (hi << 16);
}

__global__ void build_bias_kernel(const float* bz, const float* br, const float* bh,
                                  const float* bls, const float* bly, const float* blc,
                                  float* bgate, float* bsum) {
  int i = blockIdx.x * 256 + threadIdx.x;
  if (i < 2048) {
    float v = 0.0f;
    if (i < 512)       v = bz[i];
    else if (i < 1024) v = br[i - 512];
    else if (i >= 1536) v = bh[i - 1536];
    bgate[i] = v;  // layout: z | r | (s@Whh gets 0) | hrest
  }
  if (i < 1024) bsum[i] = bls[i] + bly[i] + blc[i];
}

__global__ void embed_kernel(const int* __restrict__ ys, const float* __restrict__ emb,
                             bf16_t* __restrict__ ysE) {
  int row = blockIdx.x;                 // 0..T*B-1
  int tok = ys[row];
  for (int d = threadIdx.x; d < 512; d += 256)
    ysE[(size_t)row * 512 + d] = f2bf(emb[(size_t)tok * 512 + d]);
}

__global__ void init_state_kernel(const float* __restrict__ s0,
                                  float* __restrict__ sF, bf16_t* __restrict__ sB) {
  int i = blockIdx.x * 256 + threadIdx.x;  // 64*512
  float v = s0[i];
  sF[i] = v;
  sB[i] = f2bf(v);
}

// ---------------------------------------------------------------- WMMA GEMM
// C[M,N] = concat_K(A0,A1) @ Bp (+ Cin) (+ bias), optional maxout+mask epilogue.
// A: bf16 row-major panels, DMA'd into a double-buffered LDS panel with
// GLOBAL_LOAD_ASYNC_TO_LDS_B128 (ASYNCcnt pipelined against the WMMA loop).
// Bp: K-pair-packed column-major dwords (two contiguous b128 per fragment).
// Block = 256 threads = 8 waves, wave tile 16x32, block tile 64x64.
template <int EPI>
__global__ __launch_bounds__(256) void gemm_bf16_wmma(
    const bf16_t* __restrict__ A0, int lda0,
    const bf16_t* __restrict__ A1, int lda1, int ksplit,
    const uint32_t* __restrict__ Bp, int K,
    const float* __restrict__ Cin, int ldCin,
    const float* __restrict__ bias,
    float* __restrict__ Out, int ldOut,
    const float* __restrict__ mask) {
  // 2 buffers x 64 rows x 32 dwords, row stride 36 dwords (conflict-free b128)
  __shared__ alignas(16) uint32_t Alds[2][64 * 36];

  const int tid  = threadIdx.x;
  const int lane = tid & 31;
  const int wave = tid >> 5;
  const int mw = wave >> 1;        // 4 M-tiles
  const int nw = wave & 1;         // 2 N-halves
  const int hi = lane >> 4;
  const int lm = lane & 15;
  const int blockRow = blockIdx.y * 64;
  const int colBase  = blockIdx.x * 64 + nw * 32;
  const int K2 = K >> 1;

  const uint32_t aBase0 = (uint32_t)(uintptr_t)&Alds[0][0];
  const uint32_t aBase1 = (uint32_t)(uintptr_t)&Alds[1][0];

  // Issue one chunk's A panel as per-lane async DMA: 64 rows x 128B, each of
  // 256 lanes moves 2 x 16B straight into LDS (no VGPR round-trip).
  auto issue_chunk = [&](int kc, uint32_t ldsBase) {
    const bf16_t* src;
    int lds_, kb;
    if (kc < ksplit) { src = A0; lds_ = lda0; kb = kc; }
    else             { src = A1; lds_ = lda1; kb = kc - ksplit; }
#pragma unroll
    for (int i = 0; i < 2; ++i) {
      int linear = tid + i * 256;          // 512 x 16B segments
      int row = linear >> 3;
      int seg = linear & 7;
      const bf16_t* gp = &src[(size_t)(blockRow + row) * lds_ + kb + seg * 8];
      uint32_t lo = ldsBase + (uint32_t)((row * 36 + seg * 4) * 4);
      asm volatile("global_load_async_to_lds_b128 %0, %1, off"
                   :: "v"(lo), "v"((unsigned long long)(uintptr_t)gp)
                   : "memory");
    }
  };

  v8f acc0 = {};
  v8f acc1 = {};

  issue_chunk(0, aBase0);

  for (int kc = 0; kc < K; kc += 64) {
    const bool last = (kc + 64 >= K);
    if (!last) issue_chunk(kc + 64, (kc & 64) ? aBase0 : aBase1);
    // async loads complete in order: allowing the 2 just-issued next-chunk ops
    // to stay in flight guarantees the current chunk's writes have landed.
    if (!last) asm volatile("s_wait_asynccnt 0x2" ::: "memory");
    else       asm volatile("s_wait_asynccnt 0x0" ::: "memory");
    __syncthreads();

    const uint32_t* apanel = (kc & 64) ? &Alds[1][0] : &Alds[0][0];
    const uint32_t* arow = &apanel[(mw * 16 + lm) * 36];

    if (!last)  // warm L2 path for next B chunk (global_prefetch_b8)
      __builtin_prefetch(&Bp[(size_t)(colBase + lm) * K2 + ((kc + 64) >> 1)], 0, 1);

#pragma unroll
    for (int kk = 0; kk < 64; kk += 32) {
      const int kk2 = kk >> 1;
      union { uint4 q[2]; v16bf v; } af;
      // A 16x32 bf16 frag: lanes 0-15 K-pairs {0..7}; lanes 16-31 {8..15}
      af.q[0] = *(const uint4*)&arow[kk2 + hi * 8];
      af.q[1] = *(const uint4*)&arow[kk2 + hi * 8 + 4];

      const int kg2 = (kc + kk) >> 1;
      {
        const uint32_t* bcol = &Bp[(size_t)(colBase + lm) * K2 + kg2 + hi * 8];
        union { uint4 q[2]; v16bf v; } bf0;
        bf0.q[0] = *(const uint4*)&bcol[0];
        bf0.q[1] = *(const uint4*)&bcol[4];
        acc0 = __builtin_amdgcn_wmma_f32_16x16x32_bf16(false, af.v, false, bf0.v,
                                                       (short)0, acc0, false, false);
      }
      {
        const uint32_t* bcol = &Bp[(size_t)(colBase + 16 + lm) * K2 + kg2 + hi * 8];
        union { uint4 q[2]; v16bf v; } bf1;
        bf1.q[0] = *(const uint4*)&bcol[0];
        bf1.q[1] = *(const uint4*)&bcol[4];
        acc1 = __builtin_amdgcn_wmma_f32_16x16x32_bf16(false, af.v, false, bf1.v,
                                                       (short)0, acc1, false, false);
      }
    }
    __syncthreads();
  }

  // epilogue: C/D frag layout — N = lm + 16j, M = mw*16 + hi*8 + v
#pragma unroll
  for (int j = 0; j < 2; ++j) {
    v8f acc = j ? acc1 : acc0;
    int gn = colBase + j * 16 + lm;
#pragma unroll
    for (int v = 0; v < 8; ++v) {
      int gm = blockRow + mw * 16 + hi * 8 + v;
      float val = acc[v];
      if (Cin)  val += Cin[(size_t)gm * ldCin + gn];
      if (bias) val += bias[gn];
      if (EPI == 0) {
        Out[(size_t)gm * ldOut + gn] = val;
      } else {
        // maxout over adjacent column pairs + sequence mask
        float other = __shfl_xor(val, 1, 32);
        float mx = fmaxf(val, other);
        if ((lm & 1) == 0)
          Out[(size_t)gm * ldOut + (gn >> 1)] = mx * mask[gm];
      }
    }
  }
}

// ------------------------------------------------- attention score + context
// One block per batch element b. Computes e = exp(tanh(SA[b]+uh[:,b])@Wa1 + ba1),
// then c[b,:] = (sum_s e_s * xs_h[s,b,:]) / sum_s e_s, written as bf16.
__global__ __launch_bounds__(256) void attn_kernel(
    const float* __restrict__ SA, const float* __restrict__ uh,
    const float* __restrict__ xs_h, const float* __restrict__ Wa1,
    const float* __restrict__ ba1, bf16_t* __restrict__ cB) {
  int b = blockIdx.x;
  __shared__ float sc[64];
  __shared__ float ssum;
  int lane = threadIdx.x & 31;
  int wave = threadIdx.x >> 5;

  const float* sa = &SA[(size_t)b * 512];
#pragma unroll
  for (int i = 0; i < 8; ++i) {
    int s = wave * 8 + i;
    const float* u = &uh[((size_t)s * BATCH + b) * 512];
    float acc = 0.0f;
    for (int d = lane; d < 512; d += 32)
      acc += tanhf(sa[d] + u[d]) * Wa1[d];
    for (int off = 16; off; off >>= 1) acc += __shfl_xor(acc, off, 32);
    if (lane == 0) sc[s] = __expf(acc + ba1[0]);
  }
  __syncthreads();
  if (threadIdx.x < 32) {
    float v = sc[lane] + sc[lane + 32];
    for (int off = 16; off; off >>= 1) v += __shfl_xor(v, off, 32);
    if (lane == 0) ssum = v;
  }
  __syncthreads();
  float inv = 1.0f / ssum;
  for (int d = threadIdx.x; d < 1024; d += 256) {
    float acc = 0.0f;
    for (int s = 0; s < SLENC; ++s)
      acc += sc[s] * xs_h[((size_t)s * BATCH + b) * 1024 + d];
    cB[(size_t)b * 1024 + d] = f2bf(acc * inv);
  }
}

// ---------------------------------------------------------------- GRU update
// G layout per row: [0:512) z-pre, [512:1024) r-pre, [1024:1536) s@Whh, [1536:2048) hrest
__global__ void gru_kernel(const float* __restrict__ G,
                           float* __restrict__ sF, bf16_t* __restrict__ sB) {
  int idx = blockIdx.x * 256 + threadIdx.x;  // 64*512
  int b = idx >> 9;
  int d = idx & 511;
  const float* g = &G[(size_t)b * 2048];
  float z = 1.0f / (1.0f + __expf(-g[d]));
  float r = 1.0f / (1.0f + __expf(-g[512 + d]));
  float h = tanhf(g[1536 + d] + r * g[1024 + d]);
  float s = sF[idx];
  float st = (1.0f - z) * s + z * h;
  sF[idx] = st;
  sB[idx] = f2bf(st);
}

// ------------------------------------------------------------------- launch
extern "C" void kernel_launch(void* const* d_in, const int* in_sizes, int n_in,
                              void* d_out, int out_size, void* d_ws, size_t ws_size,
                              hipStream_t stream) {
  (void)in_sizes; (void)n_in; (void)out_size; (void)ws_size;
  const float* s0      = (const float*)d_in[0];
  const float* xs_h    = (const float*)d_in[1];
  const float* uh      = (const float*)d_in[2];
  const int*   ys      = (const int*)d_in[3];
  const float* ys_mask = (const float*)d_in[4];
  const float* emb     = (const float*)d_in[5];
  const float* W_sa = (const float*)d_in[6],  *b_sa = (const float*)d_in[7];
  const float* W_a1 = (const float*)d_in[8],  *b_a1 = (const float*)d_in[9];
  const float* Wxz  = (const float*)d_in[10], *Whz  = (const float*)d_in[11];
  const float* Wcz  = (const float*)d_in[12], *bz   = (const float*)d_in[13];
  const float* Wxr  = (const float*)d_in[14], *Whr  = (const float*)d_in[15];
  const float* Wcr  = (const float*)d_in[16], *br   = (const float*)d_in[17];
  const float* Wxh  = (const float*)d_in[18], *Whh  = (const float*)d_in[19];
  const float* Wch  = (const float*)d_in[20], *bh   = (const float*)d_in[21];
  const float* W_ls = (const float*)d_in[22], *b_ls = (const float*)d_in[23];
  const float* W_ly = (const float*)d_in[24], *b_ly = (const float*)d_in[25];
  const float* W_lc = (const float*)d_in[26], *b_lc = (const float*)d_in[27];
  float* out = (float*)d_out;

  // ---- workspace carve (256B aligned) ----
  uint8_t* w = (uint8_t*)d_ws;
  auto take = [&](size_t bytes) -> void* {
    uint8_t* p = w;
    w += (bytes + 255) & ~(size_t)255;
    return (void*)p;
  };
  uint32_t* WsaP = (uint32_t*)take((size_t)256 * 512 * 4);    // K=512,  N=512
  uint32_t* WgP  = (uint32_t*)take((size_t)768 * 2048 * 4);   // K=1536, N=2048
  uint32_t* WxP  = (uint32_t*)take((size_t)256 * 3072 * 4);   // K=512,  N=3072
  uint32_t* WoP  = (uint32_t*)take((size_t)768 * 1024 * 4);   // K=1536, N=1024
  bf16_t*   ysE  = (bf16_t*)take((size_t)3072 * 512 * 2);
  float*    Xpre = (float*)take((size_t)3072 * 3072 * 4);
  float*    SA   = (float*)take((size_t)64 * 512 * 4);
  float*    G    = (float*)take((size_t)64 * 2048 * 4);
  float*    sF   = (float*)take((size_t)64 * 512 * 4);
  bf16_t*   sB   = (bf16_t*)take((size_t)64 * 512 * 2);
  bf16_t*   cB   = (bf16_t*)take((size_t)64 * 1024 * 2);
  float*    bgate= (float*)take((size_t)2048 * 4);
  float*    bsum = (float*)take((size_t)1024 * 4);

  auto cdiv = [](int a, int b) { return (a + b - 1) / b; };

  // ---- one-time prep (re-run every call for determinism; negligible cost) ----
  zero_u32_kernel<<<cdiv(768 * 2048, 256), 256, 0, stream>>>(WgP, 768 * 2048);
  zero_u32_kernel<<<cdiv(256 * 3072, 256), 256, 0, stream>>>(WxP, 256 * 3072);

  auto pack = [&](const float* src, int Ks, int Ns, uint32_t* dst, int K2tot,
                  int rowOff2, int colOff) {
    int total = (Ks >> 1) * Ns;
    pack_weight_kernel<<<cdiv(total, 256), 256, 0, stream>>>(src, Ks, Ns, dst,
                                                             K2tot, rowOff2, colOff, total);
  };
  pack(W_sa, 512, 512, WsaP, 256, 0, 0);
  // gates weight: rows[s]=Whz|Whr|Whh|0 ; rows[c]=Wcz|Wcr|0|Wch
  pack(Whz, 512, 512,  WgP, 768, 0,   0);
  pack(Whr, 512, 512,  WgP, 768, 0,   512);
  pack(Whh, 512, 512,  WgP, 768, 0,   1024);
  pack(Wcz, 1024, 512, WgP, 768, 256, 0);
  pack(Wcr, 1024, 512, WgP, 768, 256, 512);
  pack(Wch, 1024, 512, WgP, 768, 256, 1536);
  // x-projection weight: Wxz|Wxr|0|Wxh|W_ly  (N=3072)
  pack(Wxz, 512, 512,   WxP, 256, 0, 0);
  pack(Wxr, 512, 512,   WxP, 256, 0, 512);
  pack(Wxh, 512, 512,   WxP, 256, 0, 1536);
  pack(W_ly, 512, 1024, WxP, 256, 0, 2048);
  // output weight: rows[s_t]=W_ls ; rows[c]=W_lc
  pack(W_ls, 512, 1024,  WoP, 768, 0,   0);
  pack(W_lc, 1024, 1024, WoP, 768, 256, 0);

  build_bias_kernel<<<8, 256, 0, stream>>>(bz, br, bh, b_ls, b_ly, b_lc, bgate, bsum);
  embed_kernel<<<TLENC * BATCH, 256, 0, stream>>>(ys, emb, ysE);
  init_state_kernel<<<128, 256, 0, stream>>>(s0, sF, sB);

  // ---- precompute all step-independent projections: Xpre = ysE @ WxP ----
  gemm_bf16_wmma<0><<<dim3(3072 / 64, 3072 / 64), 256, 0, stream>>>(
      ysE, 512, ysE, 512, /*ksplit=*/512, WxP, /*K=*/512,
      nullptr, 0, nullptr, Xpre, 3072, nullptr);

  // ---- sequential decode: 5 kernels/step, ordered on the stream ----
  for (int t = 0; t < TLENC; ++t) {
    // SA = s @ W_sa + b_sa
    gemm_bf16_wmma<0><<<dim3(8, 1), 256, 0, stream>>>(
        sB, 512, sB, 512, 512, WsaP, 512, nullptr, 0, b_sa, SA, 512, nullptr);
    // attention scores + context c (bf16)
    attn_kernel<<<dim3(BATCH), 256, 0, stream>>>(SA, uh, xs_h, W_a1, b_a1, cB);
    // gates: G = Xpre[t] + [s|c] @ WgP + bgate   (N=2048)
    gemm_bf16_wmma<0><<<dim3(32, 1), 256, 0, stream>>>(
        sB, 512, cB, 1024, /*ksplit=*/512, WgP, /*K=*/1536,
        Xpre + (size_t)t * BATCH * 3072, 3072, bgate, G, 2048, nullptr);
    // GRU elementwise update -> s_t (fp32 + bf16)
    gru_kernel<<<dim3(128), 256, 0, stream>>>(G, sF, sB);
    // logits: maxout( [s_t|c] @ WoP + Xly[t] + bsum ) * mask -> out[t]
    gemm_bf16_wmma<1><<<dim3(16, 1), 256, 0, stream>>>(
        sB, 512, cB, 1024, /*ksplit=*/512, WoP, /*K=*/1536,
        Xpre + (size_t)t * BATCH * 3072 + 2048, 3072, bsum,
        out + (size_t)t * BATCH * 512, 512, ys_mask + (size_t)t * BATCH);
  }
}
